// DownBlock_18932215841391
// MI455X (gfx1250) — matrix operand order repaired and checked by reading, not compiled, not run on hardware
//
#include <hip/hip_runtime.h>
#include <cstdint>

typedef __attribute__((ext_vector_type(16))) __bf16        v16bf;
typedef __attribute__((ext_vector_type(8)))  float         v8f;
typedef __attribute__((ext_vector_type(4)))  unsigned int  v4u;
typedef __attribute__((ext_vector_type(4)))  int           v4i;
typedef __attribute__((ext_vector_type(4)))  float         v4f;

#define B_      8
#define INC     32
#define OUTC    64
#define VF      163842
#define VC      40962
#define NROW    (VC*B_)           /* 327696 rows of the GEMM (v*8+b) */
#define NT      (NROW/16)         /* 20481 16-row tiles (exact)      */
#define NBCONV  ((NT+3)/4)        /* 5121 blocks, 4 waves/block      */
#define K1      224               /* conv1 reduction = 7*32          */
#define K1P     232               /* padded LDS row (bank-conflict-free) */
#define K2      448               /* conv2 reduction = 7*64          */
#define K2P     456
#define INV_N   (1.0f/327696.0f)
#define EPS_    1e-5f
#define NSLOPE  0.2f

union ABu { v4u q[2]; v16bf v; };

// ---- CDNA5 async global->LDS copy path (ASYNCcnt), with safe fallback ------
#define AS1 __attribute__((address_space(1)))
#define AS3 __attribute__((address_space(3)))

#if defined(__has_builtin)
#  if __has_builtin(__builtin_amdgcn_global_load_async_to_lds_b128)
#    define HAVE_ASYNC_LDS 1
#  endif
#  if __has_builtin(__builtin_amdgcn_s_wait_asynccnt)
#    define HAVE_WAIT_ASYNC 1
#  endif
#endif
#ifndef HAVE_ASYNC_LDS
#  define HAVE_ASYNC_LDS 0
#endif
#ifndef HAVE_WAIT_ASYNC
#  define HAVE_WAIT_ASYNC 0
#endif

#if HAVE_ASYNC_LDS
// 16-byte async copy, immediate offset OFF (bytes), default cache policy.
// Signature (from compiler diagnostic): (v4i AS1*, v4i AS3*, imm offset, imm cpol)
#  define ASYNC_CP16(gptr, ldsptr, OFF)                                       \
     __builtin_amdgcn_global_load_async_to_lds_b128(                          \
         (AS1 v4i*)(gptr), (AS3 v4i*)(ldsptr), (OFF), 0)
#  if HAVE_WAIT_ASYNC
#    define WAIT_STAGE() __builtin_amdgcn_s_wait_asynccnt(0)
#  else
#    define WAIT_STAGE() asm volatile("s_wait_asynccnt 0x0" ::: "memory")
#  endif
#else
#  define WAIT_STAGE() asm volatile("s_wait_dscnt 0x0" ::: "memory")
#endif

// ---------------------------------------------------------------------------
// Repack weights: Wc[o][k*C + c] = W[o][c][k], bf16. Wc1 rows padded to K1P.
// ---------------------------------------------------------------------------
__global__ __launch_bounds__(256) void prep_weights(
    const float* __restrict__ W1, const float* __restrict__ W2,
    __bf16* __restrict__ Wc1, __bf16* __restrict__ Wc2)
{
  int i = blockIdx.x * 256 + threadIdx.x;
  if (i < OUTC * K1) {
    int o = i / K1, kk = i - o * K1;
    int k = kk >> 5, c = kk & 31;
    Wc1[o * K1P + kk] = (__bf16)W1[(o * INC + c) * 7 + k];
  }
  if (i < OUTC * (K1P - K1)) {           // zero the pad tail
    int o = i >> 3;
    Wc1[o * K1P + K1 + (i & 7)] = (__bf16)0.0f;
  }
  if (i < OUTC * K2) {
    int o = i / K2, kk = i - o * K2;
    int k = kk >> 6, c = kk & 63;
    Wc2[o * K2 + kk] = (__bf16)W2[(o * OUTC + c) * 7 + k];
  }
}

// ---------------------------------------------------------------------------
// Pool: xp_t[b][v][c] = bf16( mean_k x[b][c][dni[di[v]][k]] ), c contiguous.
// ---------------------------------------------------------------------------
__global__ __launch_bounds__(256) void pool_kernel(
    const float* __restrict__ x, const int* __restrict__ dni,
    const int* __restrict__ di, __bf16* __restrict__ xp)
{
  int r = blockIdx.x * 256 + threadIdx.x;
  if (r >= NROW) return;
  int b = r / VC, v = r - b * VC;
  int base = di[v] * 7;
  float acc[INC];
#pragma unroll
  for (int c = 0; c < INC; ++c) acc[c] = 0.0f;
  for (int k = 0; k < 7; ++k) {
    int n = dni[base + k];
    const float* xr = x + (size_t)b * INC * VF + n;
#pragma unroll
    for (int c = 0; c < INC; ++c) acc[c] += xr[(size_t)c * VF];
  }
  union { __bf16 hh[INC]; v4u q[4]; } pk;
#pragma unroll
  for (int c = 0; c < INC; ++c) pk.hh[c] = (__bf16)(acc[c] * (1.0f / 7.0f));
  v4u* dst = (v4u*)(xp + ((size_t)r << 5));
  dst[0] = pk.q[0]; dst[1] = pk.q[1]; dst[2] = pk.q[2]; dst[3] = pk.q[3];
}

// ---------------------------------------------------------------------------
// Conv1: rows R = v*8+b, D[R][o] = sum_ck xp[b][cni[v][k]][c] * W1[o][ck] + b1
// 4 waves/block, each wave one 16x64 tile: 7 K-steps x 4 wmma_f32_16x16x32_bf16
// Weights + gathered A tile staged via async global->LDS when available.
// Emits raw h1 + deterministic per-block channel partial sums/sumsq.
// ---------------------------------------------------------------------------
__global__ __launch_bounds__(128) void conv1_wmma(
    const __bf16* __restrict__ xp, const __bf16* __restrict__ Wc,
    const int* __restrict__ cni, const float* __restrict__ bias,
    float* __restrict__ h1, float* __restrict__ bstat)
{
  __shared__ __bf16 sW[OUTC * K1P];
  __shared__ __bf16 sA[4][16 * K1P];
  __shared__ float  sred[4][128];

  const int tid = threadIdx.x;
  { // weights -> LDS (straight copy, already padded/packed)
    const v4u* src = (const v4u*)Wc;
    v4u* dst = (v4u*)sW;
    for (int i = tid; i < (OUTC * K1P) / 8; i += 128) {
#if HAVE_ASYNC_LDS
      ASYNC_CP16(src + i, dst + i, 0);
#else
      dst[i] = src[i];
#endif
    }
  }
#if HAVE_ASYNC_LDS
  WAIT_STAGE();
#endif
  __syncthreads();

  const int w = tid >> 5, lane = tid & 31;
  const int tile = blockIdx.x * 4 + w;
  const bool active = tile < NT;
  __bf16* sAw = sA[w];
  const int hf = lane >> 4, m = lane & 15;

  v8f acc0 = {}, acc1 = {}, acc2 = {}, acc3 = {};

  if (active) {
    const int R0 = tile * 16;
    const int v0 = R0 >> 3;                      // tile covers v0, v0+1
    // stage A tile: 112 chunks (row 0..15 x k 0..6) of 32 contiguous bf16
    for (int cc = lane; cc < 112; cc += 32) {
      int row = cc / 7, k = cc - row * 7;
      int v = v0 + (row >> 3), b = row & 7;
      int n = cni[v * 7 + k];
      const v4u* src = (const v4u*)(xp + ((size_t)(b * VC + n) << 5));
      v4u* dst = (v4u*)(sAw + row * K1P + k * 32);
#if HAVE_ASYNC_LDS
      ASYNC_CP16(src, dst, 0);
      ASYNC_CP16(src, dst, 16);
      ASYNC_CP16(src, dst, 32);
      ASYNC_CP16(src, dst, 48);
#else
      dst[0] = src[0]; dst[1] = src[1]; dst[2] = src[2]; dst[3] = src[3];
#endif
    }
    WAIT_STAGE();

    const __bf16* arow = sAw + m * K1P + 8 * hf;
#pragma unroll
    for (int s = 0; s < 7; ++s) {
      ABu ua;
      ua.q[0] = *(const v4u*)(arow + 32 * s);
      ua.q[1] = *(const v4u*)(arow + 32 * s + 16);
      const __bf16* wb = sW + m * K1P + 32 * s + 8 * hf;
      ABu u0, u1, u2, u3;
      u0.q[0] = *(const v4u*)(wb + 0 * 16 * K1P); u0.q[1] = *(const v4u*)(wb + 0 * 16 * K1P + 16);
      u1.q[0] = *(const v4u*)(wb + 1 * 16 * K1P); u1.q[1] = *(const v4u*)(wb + 1 * 16 * K1P + 16);
      u2.q[0] = *(const v4u*)(wb + 2 * 16 * K1P); u2.q[1] = *(const v4u*)(wb + 2 * 16 * K1P + 16);
      u3.q[0] = *(const v4u*)(wb + 3 * 16 * K1P); u3.q[1] = *(const v4u*)(wb + 3 * 16 * K1P + 16);
      acc0 = __builtin_amdgcn_wmma_f32_16x16x32_bf16(false, ua.v, false, u0.v, (short)0, acc0, false, false);
      acc1 = __builtin_amdgcn_wmma_f32_16x16x32_bf16(false, ua.v, false, u1.v, (short)0, acc1, false, false);
      acc2 = __builtin_amdgcn_wmma_f32_16x16x32_bf16(false, ua.v, false, u2.v, (short)0, acc2, false, false);
      acc3 = __builtin_amdgcn_wmma_f32_16x16x32_bf16(false, ua.v, false, u3.v, (short)0, acc3, false, false);
    }

    float* outp = h1 + (size_t)R0 * 64;
#define EPI1(T, ACC)                                                          \
    { int col = (T) * 16 + m;                                                 \
      float bs = bias[col];                                                   \
      float s1 = 0.0f, s2 = 0.0f;                                             \
      _Pragma("unroll")                                                       \
      for (int r = 0; r < 8; ++r) {                                           \
        float val = ACC[r] + bs;                                              \
        outp[(size_t)(r + 8 * hf) * 64 + col] = val;                          \
        s1 += val; s2 += val * val;                                           \
      }                                                                       \
      s1 += __shfl_down(s1, 16, 32);                                          \
      s2 += __shfl_down(s2, 16, 32);                                          \
      if (hf == 0) { sred[w][col] = s1; sred[w][64 + col] = s2; } }
    EPI1(0, acc0) EPI1(1, acc1) EPI1(2, acc2) EPI1(3, acc3)
#undef EPI1
  } else if (lane < 16) {
#pragma unroll
    for (int t = 0; t < 4; ++t) {
      sred[w][t * 16 + lane] = 0.0f;
      sred[w][64 + t * 16 + lane] = 0.0f;
    }
  }
  __syncthreads();
  // fixed-order block combine (deterministic; no float atomics)
  float p = sred[0][tid] + sred[1][tid] + sred[2][tid] + sred[3][tid];
  bstat[(size_t)blockIdx.x * 128 + tid] = p;
}

// ---------------------------------------------------------------------------
// BN stats: fixed-order reduce over block partials -> per-channel scale/shift
// ---------------------------------------------------------------------------
__global__ __launch_bounds__(128) void stats_kernel(
    const float* __restrict__ bstat, int nblk,
    const float* __restrict__ gamma, const float* __restrict__ beta,
    float* __restrict__ ss)
{
  __shared__ float tot[128];
  int i = threadIdx.x;
  float s = 0.0f;
  for (int b = 0; b < nblk; ++b) s += bstat[(size_t)b * 128 + i];
  tot[i] = s;
  __syncthreads();
  if (i < 64) {
    float mean = tot[i] * INV_N;
    float var  = tot[64 + i] * INV_N - mean * mean;
    float sc   = gamma[i] * rsqrtf(var + EPS_);
    ss[i]      = sc;
    ss[64 + i] = fmaf(-mean, sc, beta[i]);
  }
}

// ---------------------------------------------------------------------------
// Conv2: same GEMM, K=448; BN1+LeakyReLU fused into the A-tile gather
// (transform runs in VGPRs, so the staging stays on the VALU/ds path).
// Weights stream from L2 (57 KB, hot); LDS budget goes to the 16x456 A tile.
// ---------------------------------------------------------------------------
__global__ __launch_bounds__(128) void conv2_wmma(
    const float* __restrict__ h1, const __bf16* __restrict__ Wc2,
    const int* __restrict__ cni, const float* __restrict__ ss,
    const float* __restrict__ bias, float* __restrict__ h2,
    float* __restrict__ bstat)
{
  __shared__ __bf16 sA[4][16 * K2P];
  __shared__ float  sred[4][128];
  __shared__ float  sSS[128];

  const int tid = threadIdx.x;
  sSS[tid] = ss[tid];
  __syncthreads();

  const int w = tid >> 5, lane = tid & 31;
  const int tile = blockIdx.x * 4 + w;
  const bool active = tile < NT;
  __bf16* sAw = sA[w];
  const int hf = lane >> 4, m = lane & 15;

  v8f acc0 = {}, acc1 = {}, acc2 = {}, acc3 = {};

  if (active) {
    const int R0 = tile * 16;
    const int v0 = R0 >> 3;
    // stage A: per chunk read 64 f32 of h1 row, apply BN1+lrelu, cvt bf16
    for (int cc = lane; cc < 112; cc += 32) {
      int row = cc / 7, k = cc - row * 7;
      int v = v0 + (row >> 3), b = row & 7;
      int n = cni[v * 7 + k];
      const v4f* src = (const v4f*)(h1 + ((size_t)(n * 8 + b) << 6));
      v4u* dst = (v4u*)(sAw + row * K2P + k * 64);
#pragma unroll
      for (int j = 0; j < 8; ++j) {
        v4f f0 = src[2 * j], f1 = src[2 * j + 1];
        union { __bf16 hh[8]; v4u q; } pk;
#pragma unroll
        for (int e = 0; e < 4; ++e) {
          int c0 = j * 8 + e, c1 = j * 8 + 4 + e;
          float y0 = fmaf(f0[e], sSS[c0], sSS[64 + c0]);
          float y1 = fmaf(f1[e], sSS[c1], sSS[64 + c1]);
          y0 = y0 > 0.0f ? y0 : NSLOPE * y0;
          y1 = y1 > 0.0f ? y1 : NSLOPE * y1;
          pk.hh[e] = (__bf16)y0; pk.hh[4 + e] = (__bf16)y1;
        }
        dst[j] = pk.q;
      }
    }
    asm volatile("s_wait_dscnt 0x0" ::: "memory");

    const __bf16* arow = sAw + m * K2P + 8 * hf;
#pragma unroll
    for (int s = 0; s < 14; ++s) {
      ABu ua;
      ua.q[0] = *(const v4u*)(arow + 32 * s);
      ua.q[1] = *(const v4u*)(arow + 32 * s + 16);
      const __bf16* wb = Wc2 + m * K2 + 32 * s + 8 * hf;
      ABu u0, u1, u2, u3;
      u0.q[0] = *(const v4u*)(wb + 0 * 16 * K2); u0.q[1] = *(const v4u*)(wb + 0 * 16 * K2 + 16);
      u1.q[0] = *(const v4u*)(wb + 1 * 16 * K2); u1.q[1] = *(const v4u*)(wb + 1 * 16 * K2 + 16);
      u2.q[0] = *(const v4u*)(wb + 2 * 16 * K2); u2.q[1] = *(const v4u*)(wb + 2 * 16 * K2 + 16);
      u3.q[0] = *(const v4u*)(wb + 3 * 16 * K2); u3.q[1] = *(const v4u*)(wb + 3 * 16 * K2 + 16);
      acc0 = __builtin_amdgcn_wmma_f32_16x16x32_bf16(false, ua.v, false, u0.v, (short)0, acc0, false, false);
      acc1 = __builtin_amdgcn_wmma_f32_16x16x32_bf16(false, ua.v, false, u1.v, (short)0, acc1, false, false);
      acc2 = __builtin_amdgcn_wmma_f32_16x16x32_bf16(false, ua.v, false, u2.v, (short)0, acc2, false, false);
      acc3 = __builtin_amdgcn_wmma_f32_16x16x32_bf16(false, ua.v, false, u3.v, (short)0, acc3, false, false);
    }

    float* outp = h2 + (size_t)R0 * 64;
#define EPI2(T, ACC)                                                          \
    { int col = (T) * 16 + m;                                                 \
      float bs = bias[col];                                                   \
      float s1 = 0.0f, s2 = 0.0f;                                             \
      _Pragma("unroll")                                                       \
      for (int r = 0; r < 8; ++r) {                                           \
        float val = ACC[r] + bs;                                              \
        outp[(size_t)(r + 8 * hf) * 64 + col] = val;                          \
        s1 += val; s2 += val * val;                                           \
      }                                                                       \
      s1 += __shfl_down(s1, 16, 32);                                          \
      s2 += __shfl_down(s2, 16, 32);                                          \
      if (hf == 0) { sred[w][col] = s1; sred[w][64 + col] = s2; } }
    EPI2(0, acc0) EPI2(1, acc1) EPI2(2, acc2) EPI2(3, acc3)
#undef EPI2
  } else if (lane < 16) {
#pragma unroll
    for (int t = 0; t < 4; ++t) {
      sred[w][t * 16 + lane] = 0.0f;
      sred[w][64 + t * 16 + lane] = 0.0f;
    }
  }
  __syncthreads();
  float p = sred[0][tid] + sred[1][tid] + sred[2][tid] + sred[3][tid];
  bstat[(size_t)blockIdx.x * 128 + tid] = p;
}

// ---------------------------------------------------------------------------
// Final: BN2 + LeakyReLU + transpose h2[(v*8+b)][o] -> out[b][o][v] via LDS
// ---------------------------------------------------------------------------
__global__ __launch_bounds__(256) void bn_act_transpose(
    const float* __restrict__ h2, const float* __restrict__ ss,
    float* __restrict__ out)
{
  __shared__ float t[128][65];
  __shared__ float lsc[64], lsh[64];
  int tid = threadIdx.x;
  if (tid < 64) { lsc[tid] = ss[tid]; lsh[tid] = ss[64 + tid]; }
  __syncthreads();
  int v0 = blockIdx.x * 128;
  int b  = blockIdx.y;
  int rem = VC - v0; if (rem > 128) rem = 128;
#pragma unroll
  for (int j = 0; j < 32; ++j) {
    int idx = tid + 256 * j;
    int r = idx >> 6, c = idx & 63;
    float val = 0.0f;
    if (r < rem) {
      float xv = h2[(((size_t)(v0 + r) * 8 + b) << 6) + c];
      float y = fmaf(xv, lsc[c], lsh[c]);
      val = y > 0.0f ? y : NSLOPE * y;
    }
    t[r][c] = val;
  }
  __syncthreads();
#pragma unroll
  for (int j = 0; j < 32; ++j) {
    int idx = tid + 256 * j;
    int o = idx >> 7, jj = idx & 127;
    if (jj < rem)
      out[((size_t)(b * 64 + o)) * VC + v0 + jj] = t[jj][o];
  }
}

// ---------------------------------------------------------------------------
extern "C" void kernel_launch(void* const* d_in, const int* in_sizes, int n_in,
                              void* d_out, int out_size, void* d_ws, size_t ws_size,
                              hipStream_t stream)
{
  (void)in_sizes; (void)n_in; (void)out_size; (void)ws_size;
  const float* x   = (const float*)d_in[0];
  const int*   cni = (const int*)d_in[1];
  const int*   dni = (const int*)d_in[2];
  const int*   di  = (const int*)d_in[3];
  const float* W1  = (const float*)d_in[4];
  const float* b1  = (const float*)d_in[5];
  const float* g1  = (const float*)d_in[6];
  const float* bt1 = (const float*)d_in[7];
  const float* W2  = (const float*)d_in[8];
  const float* b2  = (const float*)d_in[9];
  const float* g2  = (const float*)d_in[10];
  const float* bt2 = (const float*)d_in[11];
  float* out = (float*)d_out;

  char* wsb = (char*)d_ws;
  size_t off = 0;
  auto wsalloc = [&](size_t bytes) -> char* {
    char* p = wsb + off;
    off += (bytes + 255) & ~(size_t)255;
    return p;
  };
  __bf16* xp  = (__bf16*)wsalloc((size_t)NROW * INC * 2);      // 21 MB
  __bf16* Wc1 = (__bf16*)wsalloc((size_t)OUTC * K1P * 2);
  __bf16* Wc2 = (__bf16*)wsalloc((size_t)OUTC * K2 * 2);
  float*  h1  = (float*) wsalloc((size_t)NROW * OUTC * 4);     // 84 MB
  float*  h2  = (float*) wsalloc((size_t)NROW * OUTC * 4);     // 84 MB
  float*  bs1 = (float*) wsalloc((size_t)NBCONV * 128 * 4);
  float*  bs2 = (float*) wsalloc((size_t)NBCONV * 128 * 4);
  float*  ss1 = (float*) wsalloc(512);
  float*  ss2 = (float*) wsalloc(512);

  prep_weights<<<(OUTC * K2 + 255) / 256, 256, 0, stream>>>(W1, W2, Wc1, Wc2);
  pool_kernel<<<(NROW + 255) / 256, 256, 0, stream>>>(x, dni, di, xp);
  conv1_wmma<<<NBCONV, 128, 0, stream>>>(xp, Wc1, cni, b1, h1, bs1);
  stats_kernel<<<1, 128, 0, stream>>>(bs1, NBCONV, g1, bt1, ss1);
  conv2_wmma<<<NBCONV, 128, 0, stream>>>(h1, Wc2, cni, ss1, b2, h2, bs2);
  stats_kernel<<<1, 128, 0, stream>>>(bs2, NBCONV, g2, bt2, ss2);
  bn_act_transpose<<<dim3((VC + 127) / 128, B_), 256, 0, stream>>>(h2, ss2, out);
}
